// GAT_17514876633906
// MI455X (gfx1250) — matrix operand order repaired and checked by reading, not compiled, not run on hardware
//
#include <hip/hip_runtime.h>
#include <hip/hip_bf16.h>
#include <stdint.h>

// ---------------------------------------------------------------------------
// 2-layer GAT forward for MI455X (gfx1250).
//  - GEMMs: v_wmma_f32_16x16x32_f16, LDS tiles filled with
//    global_load_async_to_lds_b128 (ASYNCcnt) and read with ds_load_b128.
//  - Edge phase: wave32-per-edge coalesced gather + global_atomic_add_f32.
// ---------------------------------------------------------------------------

typedef __attribute__((ext_vector_type(16))) _Float16 v16h;
typedef __attribute__((ext_vector_type(8)))  _Float16 v8h;
typedef __attribute__((ext_vector_type(8)))  float    v8f;

#define HEADS 8
#define NEG_SLOPE 0.2f

// ---- order-preserving float<->uint encoding for atomicMax on floats --------
__device__ __forceinline__ unsigned enc_f(float f) {
    unsigned u = __float_as_uint(f);
    return (u & 0x80000000u) ? ~u : (u | 0x80000000u);
}
__device__ __forceinline__ float dec_f(unsigned e) {
    return __uint_as_float((e & 0x80000000u) ? (e & 0x7FFFFFFFu) : ~e);
}

// ---- CDNA5 async global->LDS 16-byte copy (tracked on ASYNCcnt) ------------
// VDST = VGPR with LDS byte offset (flat addr low 32 bits == LDS offset,
// ISA 10.2 aperture mapping), VADDR = 64-bit global address.
__device__ __forceinline__ void async_ld16(unsigned lds_off, const void* gptr) {
    asm volatile("global_load_async_to_lds_b128 %0, %1, off"
                 :: "v"(lds_off), "v"((unsigned long long)(uintptr_t)gptr)
                 : "memory");
}
__device__ __forceinline__ void wait_async0() {
    asm volatile("s_wait_asynccnt 0x0" ::: "memory");
}
__device__ __forceinline__ unsigned lds_off32(const void* p) {
    return (unsigned)(uintptr_t)p;   // low 32 bits of flat LDS address
}

// ---------------------------------------------------------------------------
// GEMM: C[M x NC] = A[M x K] * Wt^T, A row-major f16, Wt col-major f16
// (Wt[c*K + k] = W[k*NC + c]).  Block = 256 thr = 8 waves; wave owns one
// 16x16 C tile; grid = (M/16, NC/128).  K staged in 64-wide LDS chunks via
// async b128 copies; fragments are contiguous 16B runs -> ds_load_b128.
// ---------------------------------------------------------------------------
template <int K, int NC>
__global__ __launch_bounds__(256)
void gemm_wmma(const _Float16* __restrict__ A, const _Float16* __restrict__ Wt,
               float* __restrict__ C, int M)
{
    __shared__ __align__(16) _Float16 sA[16 * 64];    //  2 KB, [row][k]
    __shared__ __align__(16) _Float16 sB[128 * 64];   // 16 KB, [col][k]
    const int tid  = threadIdx.x;
    const int lane = tid & 31;
    const int wid  = tid >> 5;
    const int rowBase = blockIdx.x * 16;
    const int colBase = blockIdx.y * 128;
    const unsigned sAo = lds_off32(sA);
    const unsigned sBo = lds_off32(sB);

    v8f acc = {};
    for (int k0 = 0; k0 < K; k0 += 64) {
        // ---- async-stage A chunk: 16 rows x 64 k = 128 x 16B granules ----
        if (tid < 128) {
            int row = tid >> 3, kb = (tid & 7) << 3;
            int gr = rowBase + row; if (gr >= M) gr = M - 1;
            async_ld16(sAo + tid * 16, A + (size_t)gr * K + k0 + kb);
        }
        // ---- async-stage B chunk: 128 cols x 64 k = 1024 x 16B granules ----
        #pragma unroll
        for (int g = tid; g < 1024; g += 256) {
            int col = g >> 3, kb = (g & 7) << 3;
            async_ld16(sBo + g * 16, Wt + (size_t)(colBase + col) * K + k0 + kb);
        }
        wait_async0();
        __syncthreads();

        const int sel = lane >> 4;               // half-wave select
        const int am  = lane & 15;               // A row within tile
        const int bn  = (lane & 15) + wid * 16;  // B column within panel
        #pragma unroll
        for (int kk = 0; kk < 64; kk += 32) {
            // A fragment (ISA 7.12.2): K = sel*8 + i  and  16 + sel*8 + i
            v8h a0 = *(const v8h*)(sA + am * 64 + kk + sel * 8);
            v8h a1 = *(const v8h*)(sA + am * 64 + kk + 16 + sel * 8);
            // B fragment: per-lane column bn, K = sel*8 + i  and  16 + sel*8 + i
            v8h b0 = *(const v8h*)(sB + bn * 64 + kk + sel * 8);
            v8h b1 = *(const v8h*)(sB + bn * 64 + kk + 16 + sel * 8);
            v16h af = __builtin_shufflevector(a0, a1, 0,1,2,3,4,5,6,7,
                                              8,9,10,11,12,13,14,15);
            v16h bf = __builtin_shufflevector(b0, b1, 0,1,2,3,4,5,6,7,
                                              8,9,10,11,12,13,14,15);
            acc = __builtin_amdgcn_wmma_f32_16x16x32_f16(
                false, af, false, bf, (short)0, acc, false, false);
        }
        __syncthreads();
    }

    // C fragment: vgpr r -> row r (lanes 0-15) / row 8+r (lanes 16-31)
    const int col  = colBase + wid * 16 + (lane & 15);
    const int rsel = (lane >> 4) * 8;
    #pragma unroll
    for (int r = 0; r < 8; ++r) {
        int grow = rowBase + r + rsel;
        if (grow < M) C[(size_t)grow * NC + col] = acc[r];
    }
}

// ---------------------------------------------------------------------------
// f32 -> f16 elementwise (activations)
// ---------------------------------------------------------------------------
__global__ void cvt_h(const float* __restrict__ in, _Float16* __restrict__ out,
                      size_t n)
{
    size_t i = blockIdx.x * (size_t)blockDim.x + threadIdx.x;
    if (i < n) out[i] = (_Float16)in[i];
}

// f32 W[K x NC] -> f16 transposed Wt[NC x K]
__global__ void cvt_wt(const float* __restrict__ W, _Float16* __restrict__ Wt,
                       int K, int NC)
{
    int i = blockIdx.x * blockDim.x + threadIdx.x;
    if (i >= K * NC) return;
    int k = i / NC, c = i % NC;
    Wt[(size_t)c * K + k] = (_Float16)W[i];
}

// ---------------------------------------------------------------------------
// Per-(node, head) attention logits: al[n,h] = sum_c h[n,h*C+c] * a[h,c]
// ---------------------------------------------------------------------------
template <int C>
__global__ void compute_al(const float* __restrict__ h,
                           const float* __restrict__ a_src,
                           const float* __restrict__ a_dst,
                           float* __restrict__ al_s, float* __restrict__ al_d,
                           int N)
{
    int idx = blockIdx.x * blockDim.x + threadIdx.x;
    if (idx >= N * HEADS) return;
    int n = idx / HEADS, hh = idx % HEADS;
    const float* hp  = h + (size_t)n * HEADS * C + hh * C;
    const float* asv = a_src + hh * C;
    const float* adv = a_dst + hh * C;
    float ss = 0.f, sd = 0.f;
    #pragma unroll 8
    for (int c = 0; c < C; ++c) { float v = hp[c]; ss += v * asv[c]; sd += v * adv[c]; }
    al_s[idx] = ss;
    al_d[idx] = sd;
}

// ---------------------------------------------------------------------------
// Edge pass 1: segment max of leaky_relu(al_src[src] + al_dst[dst]) per dst.
// ---------------------------------------------------------------------------
__global__ void edge_max(const int* __restrict__ srcI, const int* __restrict__ dstI,
                         int E, int N,
                         const float* __restrict__ al_s, const float* __restrict__ al_d,
                         unsigned* __restrict__ menc)
{
    int e = blockIdx.x * blockDim.x + threadIdx.x;
    int Et = E + N;
    if (e >= Et) return;
    int s = (e < E) ? srcI[e] : (e - E);
    int d = (e < E) ? dstI[e] : (e - E);
    #pragma unroll
    for (int hh = 0; hh < HEADS; ++hh) {
        float v = al_s[s * HEADS + hh] + al_d[d * HEADS + hh];
        v = (v < 0.f) ? NEG_SLOPE * v : v;
        atomicMax(&menc[d * HEADS + hh], enc_f(v));
    }
}

// ---------------------------------------------------------------------------
// Edge pass 2: ex = exp(e - m[dst]); store per-edge ex; segment-sum into s.
// ---------------------------------------------------------------------------
__global__ void edge_expsum(const int* __restrict__ srcI, const int* __restrict__ dstI,
                            int E, int N,
                            const float* __restrict__ al_s, const float* __restrict__ al_d,
                            const unsigned* __restrict__ menc,
                            float* __restrict__ exbuf, float* __restrict__ ssum)
{
    int e = blockIdx.x * blockDim.x + threadIdx.x;
    int Et = E + N;
    if (e >= Et) return;
    int s = (e < E) ? srcI[e] : (e - E);
    int d = (e < E) ? dstI[e] : (e - E);
    #pragma unroll
    for (int hh = 0; hh < HEADS; ++hh) {
        float v = al_s[s * HEADS + hh] + al_d[d * HEADS + hh];
        v = (v < 0.f) ? NEG_SLOPE * v : v;
        float m  = dec_f(menc[d * HEADS + hh]);
        float ev = __expf(v - m);
        exbuf[(size_t)e * HEADS + hh] = ev;
        atomicAdd(&ssum[d * HEADS + hh], ev);
    }
}

// ---------------------------------------------------------------------------
// Edge pass 3: out[dst] += h[src] * alpha.  One wave32 per edge; lanes stride
// the F features so gathers + atomics are fully coalesced (L2-resident).
// ---------------------------------------------------------------------------
template <int F, int C>
__global__ __launch_bounds__(256)
void edge_scatter(const float* __restrict__ h, const float* __restrict__ exbuf,
                  const float* __restrict__ ssum,
                  const int* __restrict__ srcI, const int* __restrict__ dstI,
                  int E, int N, float* __restrict__ out)
{
    int wave = (int)((blockIdx.x * (size_t)blockDim.x + threadIdx.x) >> 5);
    int lane = threadIdx.x & 31;
    int Et = E + N;
    if (wave >= Et) return;
    int s = (wave < E) ? srcI[wave] : (wave - E);
    int d = (wave < E) ? dstI[wave] : (wave - E);
    #pragma unroll
    for (int f = lane; f < F; f += 32) {
        int hh = f / C;
        float alpha = exbuf[(size_t)wave * HEADS + hh] / ssum[d * HEADS + hh];
        atomicAdd(&out[(size_t)d * F + f], h[(size_t)s * F + f] * alpha);
    }
}

// ---------------------------------------------------------------------------
// Layer-1 epilogue: aggh = (f16) elu(agg + b1[col])   over [N, 256]
// (emits the f16 GEMM-2 input directly; no extra f32 pass)
// ---------------------------------------------------------------------------
__global__ void bias_elu_h(const float* __restrict__ p, const float* __restrict__ b,
                           _Float16* __restrict__ outh, size_t n)
{
    size_t i = blockIdx.x * (size_t)blockDim.x + threadIdx.x;
    if (i >= n) return;
    float v = p[i] + b[i & 255];
    v = (v > 0.f) ? v : (__expf(v) - 1.f);
    outh[i] = (_Float16)v;
}

// ---------------------------------------------------------------------------
// Final: mean over heads + b2 + log_softmax over 16 classes.
// ---------------------------------------------------------------------------
__global__ void finalize(const float* __restrict__ out2, const float* __restrict__ b2,
                         float* __restrict__ y, int N)
{
    int n = blockIdx.x * blockDim.x + threadIdx.x;
    if (n >= N) return;
    float v[16];
    #pragma unroll
    for (int c = 0; c < 16; ++c) {
        float a = 0.f;
        #pragma unroll
        for (int hh = 0; hh < HEADS; ++hh) a += out2[(size_t)n * 128 + hh * 16 + c];
        v[c] = a * (1.f / HEADS) + b2[c];
    }
    float mx = v[0];
    #pragma unroll
    for (int c = 1; c < 16; ++c) mx = fmaxf(mx, v[c]);
    float sm = 0.f;
    #pragma unroll
    for (int c = 0; c < 16; ++c) sm += __expf(v[c] - mx);
    float lse = mx + __logf(sm);
    #pragma unroll
    for (int c = 0; c < 16; ++c) y[(size_t)n * 16 + c] = v[c] - lse;
}

// ---------------------------------------------------------------------------
extern "C" void kernel_launch(void* const* d_in, const int* in_sizes, int n_in,
                              void* d_out, int out_size, void* d_ws, size_t ws_size,
                              hipStream_t stream)
{
    const float* x   = (const float*)d_in[0];
    const int*   ei  = (const int*)  d_in[1];   // [2, E] flat (src row, dst row)
    const float* W1  = (const float*)d_in[2];
    const float* as1 = (const float*)d_in[3];
    const float* ad1 = (const float*)d_in[4];
    const float* b1  = (const float*)d_in[5];
    const float* W2  = (const float*)d_in[6];
    const float* as2 = (const float*)d_in[7];
    const float* ad2 = (const float*)d_in[8];
    const float* b2  = (const float*)d_in[9];
    float* y = (float*)d_out;

    const int N = in_sizes[0] / 128;   // 100000
    const int E = in_sizes[1] / 2;     // 800000
    const int Et = E + N;              // with self loops
    const int* srcI = ei;
    const int* dstI = ei + E;

    // ---- workspace layout ----
    float* ws   = (float*)d_ws;
    float* h1   = ws;                              // N*256 f32 (reused: h2 [0,N*128), out2 [N*128,N*256))
    float* agg  = h1  + (size_t)N * 256;           // N*256 f32
    float* al_s = agg + (size_t)N * 256;           // N*8
    float* al_d = al_s + (size_t)N * HEADS;        // N*8
    unsigned* menc = (unsigned*)(al_d + (size_t)N * HEADS);   // N*8
    float* ssum = (float*)(menc + (size_t)N * HEADS);         // N*8
    float* ex   = ssum + (size_t)N * HEADS;                   // (E+N)*8
    _Float16* xh   = (_Float16*)(ex + (size_t)Et * HEADS);    // N*128 f16
    _Float16* w1t  = xh  + (size_t)N * 128;                   // 256x128 f16 (transposed)
    _Float16* w2t  = w1t + 256 * 128;                         // 128x256 f16 (transposed)
    _Float16* aggh = w2t + 128 * 256;                         // N*256 f16
    float* h2   = h1;
    float* out2 = h1 + (size_t)N * 128;

    const int mt = (N + 15) / 16;
    dim3 thr(256);

    // =============================== layer 1 ===============================
    hipMemsetAsync(agg,  0, (size_t)N * 256 * sizeof(float), stream);
    hipMemsetAsync(menc, 0, (size_t)N * HEADS * sizeof(unsigned), stream);
    hipMemsetAsync(ssum, 0, (size_t)N * HEADS * sizeof(float), stream);

    cvt_h<<<(int)(((size_t)N * 128 + 255) / 256), thr, 0, stream>>>(x, xh, (size_t)N * 128);
    cvt_wt<<<(128 * 256 + 255) / 256, thr, 0, stream>>>(W1, w1t, 128, 256);
    cvt_wt<<<(256 * 128 + 255) / 256, thr, 0, stream>>>(W2, w2t, 256, 128);

    gemm_wmma<128, 256><<<dim3(mt, 2), thr, 0, stream>>>(xh, w1t, h1, N);
    compute_al<32><<<(N * HEADS + 255) / 256, thr, 0, stream>>>(h1, as1, ad1, al_s, al_d, N);
    edge_max<<<(Et + 255) / 256, thr, 0, stream>>>(srcI, dstI, E, N, al_s, al_d, menc);
    edge_expsum<<<(Et + 255) / 256, thr, 0, stream>>>(srcI, dstI, E, N, al_s, al_d, menc, ex, ssum);
    edge_scatter<256, 32><<<(Et + 7) / 8, thr, 0, stream>>>(h1, ex, ssum, srcI, dstI, E, N, agg);
    bias_elu_h<<<(int)(((size_t)N * 256 + 255) / 256), thr, 0, stream>>>(agg, b1, aggh, (size_t)N * 256);

    // =============================== layer 2 ===============================
    gemm_wmma<256, 128><<<dim3(mt, 1), thr, 0, stream>>>(aggh, w2t, h2, N);

    hipMemsetAsync(out2, 0, (size_t)N * 128 * sizeof(float), stream);
    hipMemsetAsync(menc, 0, (size_t)N * HEADS * sizeof(unsigned), stream);
    hipMemsetAsync(ssum, 0, (size_t)N * HEADS * sizeof(float), stream);

    compute_al<16><<<(N * HEADS + 255) / 256, thr, 0, stream>>>(h2, as2, ad2, al_s, al_d, N);
    edge_max<<<(Et + 255) / 256, thr, 0, stream>>>(srcI, dstI, E, N, al_s, al_d, menc);
    edge_expsum<<<(Et + 255) / 256, thr, 0, stream>>>(srcI, dstI, E, N, al_s, al_d, menc, ex, ssum);
    edge_scatter<128, 16><<<(Et + 7) / 8, thr, 0, stream>>>(h2, ex, ssum, srcI, dstI, E, N, out2);

    finalize<<<(N + 255) / 256, thr, 0, stream>>>(out2, b2, y, N);
}
